// TemporalAttention_1082331759175
// MI455X (gfx1250) — matrix operand order repaired and checked by reading, not compiled, not run on hardware
//
#include <hip/hip_runtime.h>
#include <hip/hip_bf16.h>

// ---------------------------------------------------------------------------
// Types & WMMA helpers (CDNA5 / gfx1250, wave32)
// ---------------------------------------------------------------------------
typedef __bf16 bf16;
typedef bf16  v16bf __attribute__((ext_vector_type(16)));
typedef bf16  v8bf  __attribute__((ext_vector_type(8)));
typedef float v8f   __attribute__((ext_vector_type(8)));
typedef int   i32x4g __attribute__((vector_size(16)));  // GCC-style, matches builtin sig

#define AS1 __attribute__((address_space(1)))
#define AS3 __attribute__((address_space(3)))

#if __has_builtin(__builtin_amdgcn_global_load_async_to_lds_b128)
#define HAVE_ASYNC 1
#else
#define HAVE_ASYNC 0
#endif

#if __has_builtin(__builtin_amdgcn_tensor_load_to_lds)
#define HAVE_TDM 1
#if __has_include(<hip/amd_detail/amd_gfx1250_TDM.h>)
#define TDM_ARGS6 1   // therock/clang-23 headers -> 6-arg builtin
#else
#define TDM_ARGS6 0   // ROCm 7.2 / clang-22 -> 5-arg builtin
#endif
#else
#define HAVE_TDM 0
#endif

// A-matrix fragment (16x32 bf16): lane holds row = lane&15,
// K groups [kb..kb+7] and [kb+16..kb+23], kb = (lane<16?0:8) + K-step.
__device__ __forceinline__ v16bf ld_frag_a(const bf16* p) {
  v8bf lo = *(const v8bf*)(p);
  v8bf hi = *(const v8bf*)(p + 16);
  return __builtin_shufflevector(lo, hi, 0,1,2,3,4,5,6,7,8,9,10,11,12,13,14,15);
}
// B-matrix fragment (32x16 bf16): lane holds col = lane&15,
// K groups [kb..kb+7] and [kb+16..kb+23], kb = (lane<16?0:8) + K-step. Tile stored [n][k].
__device__ __forceinline__ v16bf ld_frag_b(const bf16* p) {
  v8bf lo = *(const v8bf*)(p);
  v8bf hi = *(const v8bf*)(p + 16);
  return __builtin_shufflevector(lo, hi, 0,1,2,3,4,5,6,7,8,9,10,11,12,13,14,15);
}

__device__ __forceinline__ v8f wmma_bf16(v16bf a, v16bf b, v8f c) {
  return __builtin_amdgcn_wmma_f32_16x16x32_bf16(false, a, false, b, (short)0, c,
                                                 false, false);
}

// --- async global->LDS (per-lane, 16B) --------------------------------------
__device__ __forceinline__ void cp16_async(void* lds, const void* g) {
#if HAVE_ASYNC
  __builtin_amdgcn_global_load_async_to_lds_b128((AS1 i32x4g*)g, (AS3 i32x4g*)lds,
                                                 0, 0);
#else
  *(uint4*)lds = *(const uint4*)g;
#endif
}
__device__ __forceinline__ void wait_async0() {
#if HAVE_ASYNC
#if __has_builtin(__builtin_amdgcn_s_wait_asynccnt)
  __builtin_amdgcn_s_wait_asynccnt(0);
#else
  asm volatile("s_wait_asynccnt 0x0" ::: "memory");
#endif
#endif
}
__device__ __forceinline__ void wait_tensor0() {
#if HAVE_TDM
#if __has_builtin(__builtin_amdgcn_s_wait_tensorcnt)
  __builtin_amdgcn_s_wait_tensorcnt(0);
#else
  asm volatile("s_wait_tensorcnt 0x0" ::: "memory");
#endif
#endif
}

__device__ __forceinline__ unsigned lds_addr_of(const void* p) {
  return (unsigned)(unsigned long long)(AS3 const void*)p;
}

// --- TDM 2D tile load: tensor[d1][d0] (elements, bf16), tile[t1][t0] --------
#if HAVE_TDM
typedef unsigned int u32x4 __attribute__((ext_vector_type(4)));
typedef int i32x8 __attribute__((ext_vector_type(8)));
typedef int i32x4 __attribute__((ext_vector_type(4)));

__device__ __forceinline__ void tdm_load_2d(const bf16* gtile, unsigned lds_byte,
                                            unsigned td0, unsigned td1,
                                            unsigned tl0, unsigned tl1,
                                            unsigned stride0) {
  unsigned long long ga = (unsigned long long)gtile;
  u32x4 g0;
  g0[0] = 1u;                                       // count=1, user descriptor
  g0[1] = lds_byte;                                 // lds_addr
  g0[2] = (unsigned)(ga & 0xFFFFFFFFull);           // global_addr[31:0]
  g0[3] = (unsigned)((ga >> 32) & 0x1FFFFFFull) | (2u << 30); // [56:32], type=2
  i32x8 g1;
  g1[0] = (int)(1u << 16);                          // data_size=1 -> 2 bytes
  g1[1] = (int)((td0 & 0xFFFFu) << 16);             // tensor_dim0[15:0]
  g1[2] = (int)(((td0 >> 16) & 0xFFFFu) | ((td1 & 0xFFFFu) << 16));
  g1[3] = (int)(((td1 >> 16) & 0xFFFFu) | (tl0 << 16)); // tile_dim0
  g1[4] = (int)(tl1 & 0xFFFFu);                     // tile_dim1 (tile_dim2=0)
  g1[5] = (int)stride0;                             // tensor_dim0_stride[31:0]
  g1[6] = 0;
  g1[7] = 0;
  i32x4 z4 = {0, 0, 0, 0};
#if TDM_ARGS6
  i32x8 z8 = {0, 0, 0, 0, 0, 0, 0, 0};
  __builtin_amdgcn_tensor_load_to_lds(g0, g1, z4, z4, z8, 0);
#else
  __builtin_amdgcn_tensor_load_to_lds(g0, g1, z4, z4, 0);
#endif
}
#endif

// Problem constants
#define NB 4
#define TT 256
#define EE 32
#define CC 512
#define HH 8
#define DD 64
#define NN (NB * EE)          // 128 sequences
#define MTOT (NN * TT)        // 32768 rows

// ---------------------------------------------------------------------------
// Kernel 0a: fp32 -> bf16 weight convert
// ---------------------------------------------------------------------------
__global__ void cvt_w_kernel(const float* __restrict__ w, bf16* __restrict__ o,
                             int nelem) {
  int i = blockIdx.x * blockDim.x + threadIdx.x;
  if (i < nelem) o[i] = (bf16)w[i];
}

// ---------------------------------------------------------------------------
// Kernel 0b: x [B,T,E,C] fp32 -> xr [(B*E), T, C] bf16 (entity fold)
// ---------------------------------------------------------------------------
__global__ void pack_x_kernel(const float* __restrict__ x, bf16* __restrict__ xbf) {
  int idx = blockIdx.x * blockDim.x + threadIdx.x;
  if (idx >= MTOT * CC) return;
  int c = idx & (CC - 1);
  int t = (idx >> 9) & (TT - 1);
  int n = idx >> 17;
  int b = n >> 5, e = n & 31;
  int src = ((b * TT + t) * EE + e) * CC + c;
  xbf[idx] = (bf16)x[src];
}

// ---------------------------------------------------------------------------
// Kernel 1: QKV projection GEMM. out[m,n] = sum_k xr[m,k] * W[n,k]
// BM=128, BN=64, BK=64, double-buffered: A-tile async-to-LDS, B-tile TDM.
// ---------------------------------------------------------------------------
__global__ __launch_bounds__(256) void qkv_kernel(const bf16* __restrict__ xbf,
                                                  const bf16* __restrict__ wbf,
                                                  bf16* __restrict__ qkv) {
  const int m0 = blockIdx.x * 128;
  const int n0 = blockIdx.y * 64;
  const int wsel = blockIdx.z;
  const bf16* W = wbf + (size_t)wsel * CC * CC;
  bf16* Out = qkv + (size_t)wsel * (size_t)MTOT * CC;

  __shared__ bf16 As[2][128][64];  // 32 KB
  __shared__ bf16 Bs[2][64][64];   // 16 KB

  const int tid = threadIdx.x;
  const int lane = tid & 31;
  const int w = tid >> 5;
  const int mi = w >> 1, ni = w & 1;
  const int l16 = lane & 15;
  const int hi8 = (lane < 16) ? 0 : 8;
  const int ka = (lane < 16) ? 0 : 8;
  const int kb = (lane < 16) ? 0 : 8;

  // tile issue: A via per-lane async, B via TDM (wave 0)
  auto issue = [&](int buf, int k0) {
#pragma unroll
    for (int i = 0; i < 4; ++i) {        // 128x64 bf16 = 1024 x 16B
      int u = tid + i * 256;
      int r = u >> 3, cg = u & 7;
      cp16_async(&As[buf][r][cg * 8], &xbf[(size_t)(m0 + r) * CC + k0 + cg * 8]);
    }
#if HAVE_TDM
    if (w == 0)
      tdm_load_2d(W + (size_t)n0 * CC + k0, lds_addr_of(&Bs[buf][0][0]),
                  CC, CC, 64, 64, CC);
#else
#pragma unroll
    for (int i = 0; i < 2; ++i) {        // 64x64 bf16 = 512 x 16B
      int u = tid + i * 256;
      int r = u >> 3, cg = u & 7;
      *(uint4*)&Bs[buf][r][cg * 8] =
          *(const uint4*)&W[(size_t)(n0 + r) * CC + k0 + cg * 8];
    }
#endif
  };

  v8f acc[2][2] = {};
  issue(0, 0);
  for (int kk = 0; kk < 8; ++kk) {
    wait_async0();
    wait_tensor0();
    __syncthreads();             // tile kk visible; tile kk-1 fully consumed
    if (kk < 7) issue((kk + 1) & 1, (kk + 1) * 64);
    const int cur = kk & 1;
#pragma unroll
    for (int ks = 0; ks < 64; ks += 32) {
      v16bf a0 = ld_frag_a(&As[cur][mi * 32 + l16][ks + ka]);
      v16bf a1 = ld_frag_a(&As[cur][mi * 32 + 16 + l16][ks + ka]);
      v16bf b0 = ld_frag_b(&Bs[cur][ni * 32 + l16][ks + kb]);
      v16bf b1 = ld_frag_b(&Bs[cur][ni * 32 + 16 + l16][ks + kb]);
      acc[0][0] = wmma_bf16(a0, b0, acc[0][0]);
      acc[0][1] = wmma_bf16(a0, b1, acc[0][1]);
      acc[1][0] = wmma_bf16(a1, b0, acc[1][0]);
      acc[1][1] = wmma_bf16(a1, b1, acc[1][1]);
    }
  }

#pragma unroll
  for (int mf = 0; mf < 2; ++mf)
#pragma unroll
    for (int nf = 0; nf < 2; ++nf)
#pragma unroll
      for (int v = 0; v < 8; ++v) {
        int row = m0 + mi * 32 + mf * 16 + v + hi8;
        int col = n0 + ni * 32 + nf * 16 + l16;
        Out[(size_t)row * CC + col] = (bf16)acc[mf][nf][v];
      }
}

// ---------------------------------------------------------------------------
// Kernel 2: flash attention. Block = (t-tile of 128, head, sequence n).
// Q via async-to-LDS; K via TDM; V manual register transpose. K/V double-buffered.
// ---------------------------------------------------------------------------
__global__ __launch_bounds__(256) void attn_kernel(const bf16* __restrict__ qkv,
                                                   bf16* __restrict__ obf) {
  const int t0 = blockIdx.x * 128;
  const int h = blockIdx.y;
  const int n = blockIdx.z;
  const float scale = 0.125f; // 1/sqrt(64)

  const bf16* Q = qkv;
  const bf16* K = qkv + (size_t)MTOT * CC;
  const bf16* V = qkv + 2ull * (size_t)MTOT * CC;

  __shared__ bf16 Qs[128][64];     // 16 KB
  __shared__ bf16 Ks[2][64][64];   // 16 KB   Ks[s][d]
  __shared__ bf16 Vts[2][64][64];  // 16 KB   Vts[d][s] (transposed)
  __shared__ bf16 Ps[8][16][64];   // 16 KB   wave-private P staging

  const int tid = threadIdx.x;
  const int lane = tid & 31;
  const int w = tid >> 5;
  const int l16 = lane & 15;
  const int hi8 = (lane < 16) ? 0 : 8;
  const int ka = (lane < 16) ? 0 : 8;
  const int kb = (lane < 16) ? 0 : 8;

  const size_t rowbase = (size_t)n * TT;

  auto issue_kv = [&](int buf, int s0) {
#if HAVE_TDM
    if (w == 0)
      tdm_load_2d(K + (rowbase + s0) * CC + h * DD,
                  lds_addr_of(&Ks[buf][0][0]), CC, MTOT, 64, 64, CC);
#else
#pragma unroll
    for (int i = 0; i < 2; ++i) {
      int u = tid + i * 256;
      int r = u >> 3, dg = u & 7;
      *(uint4*)&Ks[buf][r][dg * 8] =
          *(const uint4*)&K[(rowbase + s0 + r) * CC + h * DD + dg * 8];
    }
#endif
#pragma unroll
    for (int i = 0; i < 2; ++i) {  // V transposed into Vts[d][s]
      int u = tid + i * 256;
      int r = u >> 3, dg = u & 7;
      uint4 vv = *(const uint4*)&V[(rowbase + s0 + r) * CC + h * DD + dg * 8];
      const bf16* pv = (const bf16*)&vv;
#pragma unroll
      for (int j = 0; j < 8; ++j) Vts[buf][dg * 8 + j][r] = pv[j];
    }
  };

  // Prologue: Q tile (async) + first K/V tiles
#pragma unroll
  for (int i = 0; i < 4; ++i) {
    int u = tid + i * 256;
    int r = u >> 3, dg = u & 7;
    cp16_async(&Qs[r][dg * 8], &Q[(rowbase + t0 + r) * CC + h * DD + dg * 8]);
  }
  issue_kv(0, 0);

  v8f o_acc[4] = {};
  float m_run[8], l_run[8];
#pragma unroll
  for (int v = 0; v < 8; ++v) { m_run[v] = -3.0e38f; l_run[v] = 0.0f; }

  for (int sblk = 0; sblk < 4; ++sblk) {
    wait_async0();
    wait_tensor0();
    __syncthreads();             // tiles for sblk visible; sblk-1 consumed
    if (sblk < 3) issue_kv((sblk + 1) & 1, (sblk + 1) * 64);
    const int cur = sblk & 1;

    // S = Q * K^T  (16 rows x 64 key-cols per wave; K dim = d = 64)
    v8f s_acc[4] = {};
    v16bf aq0 = ld_frag_a(&Qs[w * 16 + l16][ka]);
    v16bf aq1 = ld_frag_a(&Qs[w * 16 + l16][32 + ka]);
#pragma unroll
    for (int si = 0; si < 4; ++si) {
      v16bf b0 = ld_frag_b(&Ks[cur][si * 16 + l16][kb]);
      v16bf b1 = ld_frag_b(&Ks[cur][si * 16 + l16][32 + kb]);
      s_acc[si] = wmma_bf16(aq0, b0, s_acc[si]);
      s_acc[si] = wmma_bf16(aq1, b1, s_acc[si]);
    }

    // Online softmax: lanes 0-15 rows v, lanes 16-31 rows v+8; xor-reduce 1..8
#pragma unroll
    for (int v = 0; v < 8; ++v) {
      float s0v = s_acc[0][v] * scale, s1v = s_acc[1][v] * scale;
      float s2v = s_acc[2][v] * scale, s3v = s_acc[3][v] * scale;
      float mx = fmaxf(fmaxf(s0v, s1v), fmaxf(s2v, s3v));
#pragma unroll
      for (int msk = 1; msk <= 8; msk <<= 1)
        mx = fmaxf(mx, __shfl_xor(mx, msk, 32));
      float mnew = fmaxf(m_run[v], mx);
      float alpha = __expf(m_run[v] - mnew);
      float p0 = __expf(s0v - mnew), p1 = __expf(s1v - mnew);
      float p2 = __expf(s2v - mnew), p3 = __expf(s3v - mnew);
      s_acc[0][v] = p0; s_acc[1][v] = p1; s_acc[2][v] = p2; s_acc[3][v] = p3;
      float rs = p0 + p1 + p2 + p3;
#pragma unroll
      for (int msk = 1; msk <= 8; msk <<= 1)
        rs += __shfl_xor(rs, msk, 32);
      m_run[v] = mnew;
      l_run[v] = l_run[v] * alpha + rs;
#pragma unroll
      for (int di = 0; di < 4; ++di) o_acc[di][v] *= alpha;
    }

    // Stage P (C/D layout -> LDS row-major), wave-private slab.
#pragma unroll
    for (int si = 0; si < 4; ++si)
#pragma unroll
      for (int v = 0; v < 8; ++v)
        Ps[w][v + hi8][si * 16 + l16] = (bf16)s_acc[si][v];
    __syncthreads();

    // O += P * V   (K dim = s = 64; B[k=s][n=d] = Vts[d][s])
    v16bf ap0 = ld_frag_a(&Ps[w][l16][ka]);
    v16bf ap1 = ld_frag_a(&Ps[w][l16][32 + ka]);
#pragma unroll
    for (int di = 0; di < 4; ++di) {
      v16bf b0 = ld_frag_b(&Vts[cur][di * 16 + l16][kb]);
      v16bf b1 = ld_frag_b(&Vts[cur][di * 16 + l16][32 + kb]);
      o_acc[di] = wmma_bf16(ap0, b0, o_acc[di]);
      o_acc[di] = wmma_bf16(ap1, b1, o_acc[di]);
    }
  }

#pragma unroll
  for (int v = 0; v < 8; ++v) {
    float inv = 1.0f / l_run[v];
    int row = t0 + w * 16 + v + hi8;
#pragma unroll
    for (int di = 0; di < 4; ++di)
      obf[(rowbase + row) * CC + h * DD + di * 16 + l16] =
          (bf16)(o_acc[di][v] * inv);
  }
}

// ---------------------------------------------------------------------------
// Kernel 3: output projection + residual (same GEMM skeleton as kernel 1).
// out[b,t,e,c] = x[b,t,e,c] + gamma * (sum_k O[m,k]*Wp[c,k] + bp[c])
// ---------------------------------------------------------------------------
__global__ __launch_bounds__(256) void oproj_kernel(
    const bf16* __restrict__ obf, const bf16* __restrict__ wpbf,
    const float* __restrict__ x, const float* __restrict__ bp,
    const float* __restrict__ gamma, float* __restrict__ out) {
  const int m0 = blockIdx.x * 128;
  const int n0 = blockIdx.y * 64;

  __shared__ bf16 As[2][128][64];
  __shared__ bf16 Bs[2][64][64];

  const int tid = threadIdx.x;
  const int lane = tid & 31;
  const int w = tid >> 5;
  const int mi = w >> 1, ni = w & 1;
  const int l16 = lane & 15;
  const int hi8 = (lane < 16) ? 0 : 8;
  const int ka = (lane < 16) ? 0 : 8;
  const int kb = (lane < 16) ? 0 : 8;

  auto issue = [&](int buf, int k0) {
#pragma unroll
    for (int i = 0; i < 4; ++i) {
      int u = tid + i * 256;
      int r = u >> 3, cg = u & 7;
      cp16_async(&As[buf][r][cg * 8], &obf[(size_t)(m0 + r) * CC + k0 + cg * 8]);
    }
#if HAVE_TDM
    if (w == 0)
      tdm_load_2d(wpbf + (size_t)n0 * CC + k0, lds_addr_of(&Bs[buf][0][0]),
                  CC, CC, 64, 64, CC);
#else
#pragma unroll
    for (int i = 0; i < 2; ++i) {
      int u = tid + i * 256;
      int r = u >> 3, cg = u & 7;
      *(uint4*)&Bs[buf][r][cg * 8] =
          *(const uint4*)&wpbf[(size_t)(n0 + r) * CC + k0 + cg * 8];
    }
#endif
  };

  v8f acc[2][2] = {};
  issue(0, 0);
  for (int kk = 0; kk < 8; ++kk) {
    wait_async0();
    wait_tensor0();
    __syncthreads();
    if (kk < 7) issue((kk + 1) & 1, (kk + 1) * 64);
    const int cur = kk & 1;
#pragma unroll
    for (int ks = 0; ks < 64; ks += 32) {
      v16bf a0 = ld_frag_a(&As[cur][mi * 32 + l16][ks + ka]);
      v16bf a1 = ld_frag_a(&As[cur][mi * 32 + 16 + l16][ks + ka]);
      v16bf b0 = ld_frag_b(&Bs[cur][ni * 32 + l16][ks + kb]);
      v16bf b1 = ld_frag_b(&Bs[cur][ni * 32 + 16 + l16][ks + kb]);
      acc[0][0] = wmma_bf16(a0, b0, acc[0][0]);
      acc[0][1] = wmma_bf16(a0, b1, acc[0][1]);
      acc[1][0] = wmma_bf16(a1, b0, acc[1][0]);
      acc[1][1] = wmma_bf16(a1, b1, acc[1][1]);
    }
  }

  const float g = gamma[0];
#pragma unroll
  for (int mf = 0; mf < 2; ++mf)
#pragma unroll
    for (int nf = 0; nf < 2; ++nf)
#pragma unroll
      for (int v = 0; v < 8; ++v) {
        int row = m0 + mi * 32 + mf * 16 + v + hi8; // m = n*T + t
        int col = n0 + ni * 32 + nf * 16 + l16;
        int nn = row >> 8, t = row & (TT - 1);
        int b = nn >> 5, e = nn & 31;
        size_t dst = ((size_t)((b * TT + t) * EE + e)) * CC + col;
        out[dst] = x[dst] + g * (acc[mf][nf][v] + bp[col]);
      }
}

// ---------------------------------------------------------------------------
// Launch
// ---------------------------------------------------------------------------
extern "C" void kernel_launch(void* const* d_in, const int* in_sizes, int n_in,
                              void* d_out, int out_size, void* d_ws,
                              size_t ws_size, hipStream_t stream) {
  const float* x = (const float*)d_in[0];
  const float* Wq = (const float*)d_in[1];
  const float* Wk = (const float*)d_in[2];
  const float* Wv = (const float*)d_in[3];
  const float* Wp = (const float*)d_in[4];
  const float* bp = (const float*)d_in[5];
  const float* gamma = (const float*)d_in[6];
  float* out = (float*)d_out;

  char* ws = (char*)d_ws;
  const size_t W_ELEMS = (size_t)CC * CC;   // 262144
  const size_t X_ELEMS = (size_t)MTOT * CC; // 16.7M
  bf16* wbf = (bf16*)ws;
  bf16* xbf = (bf16*)(ws + 4 * W_ELEMS * sizeof(bf16));
  bf16* qkv = (bf16*)((char*)xbf + X_ELEMS * sizeof(bf16));
  bf16* obf = (bf16*)((char*)qkv + 3 * X_ELEMS * sizeof(bf16));

  cvt_w_kernel<<<1024, 256, 0, stream>>>(Wq, wbf + 0 * W_ELEMS, (int)W_ELEMS);
  cvt_w_kernel<<<1024, 256, 0, stream>>>(Wk, wbf + 1 * W_ELEMS, (int)W_ELEMS);
  cvt_w_kernel<<<1024, 256, 0, stream>>>(Wv, wbf + 2 * W_ELEMS, (int)W_ELEMS);
  cvt_w_kernel<<<1024, 256, 0, stream>>>(Wp, wbf + 3 * W_ELEMS, (int)W_ELEMS);
  pack_x_kernel<<<(int)(X_ELEMS / 256), 256, 0, stream>>>(x, xbf);

  qkv_kernel<<<dim3(MTOT / 128, CC / 64, 3), 256, 0, stream>>>(xbf, wbf, qkv);
  attn_kernel<<<dim3(TT / 128, HH, NN), 256, 0, stream>>>(qkv, obf);
  oproj_kernel<<<dim3(MTOT / 128, CC / 64, 1), 256, 0, stream>>>(
      obf, wbf + 3 * W_ELEMS, x, bp, gamma, out);
}
